// Seq2Seq_4930622456243
// MI455X (gfx1250) — matrix operand order repaired and checked by reading, not compiled, not run on hardware
//
#include <hip/hip_runtime.h>

// ---------------------------------------------------------------------------
// Seq2Seq (tanh-RNN encoder + attention decoder) for MI455X gfx1250.
// All GEMMs run through v_wmma_f32_16x16x32_bf16 (fp32 accumulate).
// Weights are converted fp32 -> bf16 per launch; the dominant
// 2016x32000x2048 output projection is hoisted out of the decoder scan and
// executed as one WMMA GEMM (bf16 h2o_W = 131 MB, L2-resident).
// GEMM tiles are moved global -> LDS with GLOBAL_LOAD_ASYNC_TO_LDS_B128
// (ASYNCcnt-tracked, zero VGPR staging -> no scratch spills), double-buffered
// with one s_barrier per 64-wide K slab; prefetch overlaps the WMMA burst.
// ---------------------------------------------------------------------------

typedef __attribute__((ext_vector_type(16))) __bf16 v16bf;
typedef __attribute__((ext_vector_type(8)))  float  v8f;

typedef unsigned short u16;

static constexpr int BN   = 32;      // batch
static constexpr int LS   = 128;     // source length
static constexpr int LT   = 64;      // target length
static constexpr int TDEC = LT - 1;  // 63 decoder steps
static constexpr int E    = 512;
static constexpr int H    = 1024;
static constexpr int V    = 32000;
static constexpr int M2   = BN * TDEC;  // 2016 output rows

__device__ __forceinline__ u16 f2bf(float x) {
    unsigned u = __float_as_uint(x);
    u = u + 0x7FFFu + ((u >> 16) & 1u);   // round-to-nearest-even
    return (u16)(u >> 16);
}

// Async 16B copy: global (per-lane 64-bit addr) -> LDS (per-lane 32-bit addr).
// LDS byte address = low 32 bits of the generic pointer (ISA flat->LDS map).
__device__ __forceinline__ void async_copy_b128(const u16* gsrc, u16* ldst) {
    const unsigned laddr = (unsigned)(size_t)ldst;
    asm volatile("global_load_async_to_lds_b128 %0, %1, off"
                 :: "v"(laddr), "v"(gsrc) : "memory");
}
__device__ __forceinline__ void async_wait0() {
    asm volatile("s_wait_asynccnt 0" ::: "memory");
}

struct bf16frag { uint4 lo; uint4 hi; };   // 32 bytes == v16bf

// ---------------------------------------------------------------------------
// Generic WMMA GEMM:  Cout[M,N] = act( A[M,K](bf16) * W[N,K](bf16)^T
//                                       + bias[N] (+ Cin[M,N]) )
// Block: 256 threads = 8 waves arranged WM x WN; wave tile = 16 x 64.
// Block tile: (WM*16) x (WN*64).  K must be a multiple of 64.
// N must be a multiple of WN*64.  MGUARD: OOB A rows are CLAMPED to row 0
// (their D rows are masked at the store), so async loads stay branch-free.
// ---------------------------------------------------------------------------
template<int WM, int WN, bool TANH, bool HASCIN, bool BF16OUT, bool MGUARD>
__global__ __launch_bounds__(256) void wmma_gemm(
    const u16*  __restrict__ A,   int lda,      // bf16, row-major, rows x K
    const u16*  __restrict__ W,                 // bf16, N x K row-major
    const float* __restrict__ bias,             // [N]
    const float* __restrict__ Cin, int ldcin,   // optional addend
    float*      __restrict__ Cout, int ldc,
    u16*        __restrict__ Bout, int ldbo,    // optional bf16 mirror
    int M, int N, int K)
{
    constexpr int KS = 64;                 // K slab (u16 elements)
    constexpr int CA = (WM * 16 * 8) / 256;   // 16B chunks per thread (A)
    constexpr int CB = (WN * 64 * 8) / 256;   // 16B chunks per thread (B)

    __shared__ __align__(16) u16 As[2][WM * 16 * KS];
    __shared__ __align__(16) u16 Bs[2][WN * 64 * KS];

    const int tid  = threadIdx.x;
    const int wave = tid >> 5;
    const int lane = tid & 31;
    const int wm   = wave % WM;
    const int wn   = wave / WM;
    const int hi   = lane >> 4;
    const int lo   = lane & 15;

    const int mBlock = blockIdx.y * (WM * 16);
    const int nBlock = blockIdx.x * (WN * 64);
    const int nWave  = nBlock + wn * 64;

    v8f acc[4];
#pragma unroll
    for (int i = 0; i < 4; ++i)
#pragma unroll
        for (int v = 0; v < 8; ++v) acc[i][v] = 0.0f;

    // ---- prologue: async-copy slab 0 into LDS buffer 0 ----
#pragma unroll
    for (int i = 0; i < CA; ++i) {
        const int c = tid + i * 256;
        int gm = mBlock + (c >> 3);
        if (MGUARD) gm = (gm < M) ? gm : 0;          // clamp; masked at store
        async_copy_b128(A + (size_t)gm * lda + (c & 7) * 8,
                        &As[0][(c >> 3) * KS + (c & 7) * 8]);
    }
#pragma unroll
    for (int i = 0; i < CB; ++i) {
        const int c = tid + i * 256;
        async_copy_b128(W + (size_t)(nBlock + (c >> 3)) * K + (c & 7) * 8,
                        &Bs[0][(c >> 3) * KS + (c & 7) * 8]);
    }
    async_wait0();
    __syncthreads();

    const int S = K / KS;
    const int rowA = wm * 16 + lo;
    int buf = 0;
    for (int s = 0; s < S; ++s) {
        // ---- async-prefetch next slab into the other buffer ----
        if (s + 1 < S) {
            const int kn = (s + 1) * KS;
#pragma unroll
            for (int i = 0; i < CA; ++i) {
                const int c = tid + i * 256;
                int gm = mBlock + (c >> 3);
                if (MGUARD) gm = (gm < M) ? gm : 0;
                async_copy_b128(A + (size_t)gm * lda + kn + (c & 7) * 8,
                                &As[buf ^ 1][(c >> 3) * KS + (c & 7) * 8]);
            }
#pragma unroll
            for (int i = 0; i < CB; ++i) {
                const int c = tid + i * 256;
                async_copy_b128(
                    W + (size_t)(nBlock + (c >> 3)) * K + kn + (c & 7) * 8,
                    &Bs[buf ^ 1][(c >> 3) * KS + (c & 7) * 8]);
            }
        }

        // ---- compute on current LDS buffer ----
#pragma unroll
        for (int kk = 0; kk < 2; ++kk) {
            // A fragment: K = {8hi..8hi+7, 8hi+16..8hi+23} within 32-K half
            bf16frag fa;
            fa.lo = *reinterpret_cast<const uint4*>(
                &As[buf][rowA * KS + kk * 32 + hi * 8]);
            fa.hi = *reinterpret_cast<const uint4*>(
                &As[buf][rowA * KS + kk * 32 + 16 + hi * 8]);
            const v16bf av = __builtin_bit_cast(v16bf, fa);

            v16bf bv[4];
#pragma unroll
            for (int nt = 0; nt < 4; ++nt) {
                const int rowB = wn * 64 + nt * 16 + lo;
                bf16frag fb;
                fb.lo = *reinterpret_cast<const uint4*>(
                    &Bs[buf][rowB * KS + kk * 32 + hi * 8]);
                fb.hi = *reinterpret_cast<const uint4*>(
                    &Bs[buf][rowB * KS + kk * 32 + 16 + hi * 8]);
                bv[nt] = __builtin_bit_cast(v16bf, fb);
            }
#pragma unroll
            for (int nt = 0; nt < 4; ++nt)
                acc[nt] = __builtin_amdgcn_wmma_f32_16x16x32_bf16(
                    false, av, false, bv[nt], (short)0, acc[nt], false, false);
        }

        // my async prefetch must be in LDS before anyone computes on it
        async_wait0();
        __syncthreads();
        buf ^= 1;
    }

    // Epilogue: C/D layout -> m = 8*hi + v, n = lo (per VGPR v)
#pragma unroll
    for (int nt = 0; nt < 4; ++nt) {
        const int n = nWave + nt * 16 + lo;
        const float bval = bias[n];
#pragma unroll
        for (int v = 0; v < 8; ++v) {
            const int m = mBlock + wm * 16 + hi * 8 + v;
            if (!MGUARD || m < M) {
                float val = acc[nt][v] + bval;
                if constexpr (HASCIN) val += Cin[(size_t)m * ldcin + n];
                if constexpr (TANH)   val = tanhf(val);
                Cout[(size_t)m * ldc + n] = val;
                if constexpr (BF16OUT) Bout[(size_t)m * ldbo + n] = f2bf(val);
            }
        }
    }
}

// ---------------------------------------------------------------------------
// Small helper kernels
// ---------------------------------------------------------------------------
__global__ void f32_to_bf16_kernel(const float* __restrict__ in,
                                   u16* __restrict__ out, size_t n) {
    size_t i = (size_t)blockIdx.x * blockDim.x + threadIdx.x;
    size_t stride = (size_t)gridDim.x * blockDim.x;
    for (; i < n; i += stride) out[i] = f2bf(in[i]);
}

__global__ void fill_zero_u16_kernel(u16* __restrict__ out, size_t n) {
    size_t i = (size_t)blockIdx.x * blockDim.x + threadIdx.x;
    size_t stride = (size_t)gridDim.x * blockDim.x;
    for (; i < n; i += stride) out[i] = 0;
}

// enc embedding gather: out[r*512+e] = bf16(table[idx[r]*512+e]), r in [0,4096)
__global__ void enc_gather_kernel(const int* __restrict__ idx,
                                  const float* __restrict__ table,
                                  u16* __restrict__ out, size_t total) {
    size_t i = (size_t)blockIdx.x * blockDim.x + threadIdx.x;
    size_t stride = (size_t)gridDim.x * blockDim.x;
    for (; i < total; i += stride) {
        size_t r = i >> 9, e = i & 511;
        out[i] = f2bf(table[((size_t)idx[r] << 9) + e]);
    }
}

// dec embedding gather for step t into xbuf[:, 0:512] (row stride E+H=1536)
__global__ void dec_gather_kernel(const int* __restrict__ tgt,
                                  const float* __restrict__ table,
                                  u16* __restrict__ xbuf, int t) {
    int i = blockIdx.x * blockDim.x + threadIdx.x;   // 32*512
    int n = i >> 9, e = i & 511;
    int idx = tgt[n * LT + t];
    xbuf[(size_t)n * (E + H) + e] = f2bf(table[((size_t)idx << 9) + e]);
}

// scores[r] = dot(energy[r, :], h[r/128, :])     r = n*128 + l
__global__ __launch_bounds__(256) void scores_kernel(
    const float* __restrict__ energy, const float* __restrict__ h, int ldh,
    float* __restrict__ scores) {
    __shared__ float red[256];
    const int r = blockIdx.x, n = r >> 7, tid = threadIdx.x;
    const float* erow = energy + (size_t)r * H;
    const float* hrow = h + (size_t)n * ldh;
    float acc = 0.f;
    for (int j = tid; j < H; j += 256) acc += erow[j] * hrow[j];
    red[tid] = acc; __syncthreads();
    for (int s = 128; s > 0; s >>= 1) {
        if (tid < s) red[tid] += red[tid + s];
        __syncthreads();
    }
    if (tid == 0) scores[r] = red[0];
}

// masked softmax over l=128 per batch row
__global__ __launch_bounds__(128) void softmax_kernel(
    const float* __restrict__ scores, const int* __restrict__ src,
    float* __restrict__ attw) {
    __shared__ float sm[128];
    const int n = blockIdx.x, l = threadIdx.x;
    float s = scores[n * LS + l];
    if (src[n * LS + l] == 0) s = -1.0e9f;   // PAD mask
    sm[l] = s; __syncthreads();
    for (int st = 64; st > 0; st >>= 1) {
        if (l < st) sm[l] = fmaxf(sm[l], sm[l + st]);
        __syncthreads();
    }
    const float mx = sm[0]; __syncthreads();
    const float e = expf(s - mx);
    sm[l] = e; __syncthreads();
    for (int st = 64; st > 0; st >>= 1) {
        if (l < st) sm[l] += sm[l + st];
        __syncthreads();
    }
    attw[n * LS + l] = e / sm[0];
}

// ctx[n,h] = sum_l attw[n,l] * enc_hs[n,l,h]; write bf16 into X and xbuf
__global__ __launch_bounds__(256) void ctx_kernel(
    const float* __restrict__ attw, const float* __restrict__ ehs,
    u16* __restrict__ X, u16* __restrict__ xbuf, int t) {
    const int n = blockIdx.x;
    const int h = blockIdx.y * 256 + threadIdx.x;
    float acc = 0.f;
    for (int l = 0; l < LS; ++l)
        acc += attw[n * LS + l] * ehs[((size_t)(n * LS + l)) * H + h];
    const u16 b = f2bf(acc);
    X[((size_t)(n * TDEC + t)) * (2 * H) + H + h] = b;   // final-GEMM A rows
    xbuf[(size_t)n * (E + H) + E + h] = b;               // decoder GEMM1 input
}

// in-place log_softmax over rows of 32000
__global__ __launch_bounds__(256) void log_softmax_kernel(float* __restrict__ out) {
    __shared__ float red[256];
    float* row = out + (size_t)blockIdx.x * V;
    const int tid = threadIdx.x;
    float mx = -3.4e38f;
    for (int j = tid; j < V; j += 256) mx = fmaxf(mx, row[j]);
    red[tid] = mx; __syncthreads();
    for (int s = 128; s > 0; s >>= 1) {
        if (tid < s) red[tid] = fmaxf(red[tid], red[tid + s]);
        __syncthreads();
    }
    mx = red[0]; __syncthreads();
    float sum = 0.f;
    for (int j = tid; j < V; j += 256) sum += expf(row[j] - mx);
    red[tid] = sum; __syncthreads();
    for (int s = 128; s > 0; s >>= 1) {
        if (tid < s) red[tid] += red[tid + s];
        __syncthreads();
    }
    const float lse = mx + logf(red[0]);
    for (int j = tid; j < V; j += 256) row[j] -= lse;
}

// ---------------------------------------------------------------------------
// Host orchestration
// ---------------------------------------------------------------------------
extern "C" void kernel_launch(void* const* d_in, const int* in_sizes, int n_in,
                              void* d_out, int out_size, void* d_ws, size_t ws_size,
                              hipStream_t stream) {
    (void)in_sizes; (void)n_in; (void)out_size; (void)ws_size;

    const int*   src       = (const int*)  d_in[0];
    const int*   tgt       = (const int*)  d_in[1];
    const float* enc_embed = (const float*)d_in[2];
    const float* enc_Wih   = (const float*)d_in[3];
    const float* enc_bih   = (const float*)d_in[4];
    const float* enc_Whh   = (const float*)d_in[5];
    const float* enc_bhh   = (const float*)d_in[6];
    const float* dec_embed = (const float*)d_in[7];
    const float* attn_W    = (const float*)d_in[8];
    const float* attn_b    = (const float*)d_in[9];
    const float* dec_Wih   = (const float*)d_in[10];
    const float* dec_bih   = (const float*)d_in[11];
    const float* dec_Whh   = (const float*)d_in[12];
    const float* dec_bhh   = (const float*)d_in[13];
    const float* h2o_W     = (const float*)d_in[14];
    const float* h2o_b     = (const float*)d_in[15];
    float* out = (float*)d_out;

    // ---- workspace carve-out (256B aligned) ----
    char* ws = (char*)d_ws;
    size_t off = 0;
    auto alloc = [&](size_t bytes) -> void* {
        void* p = ws + off;
        off = (off + bytes + 255) & ~(size_t)255;
        return p;
    };
    u16*   w_encWih = (u16*)  alloc((size_t)H * E * 2);
    u16*   w_encWhh = (u16*)  alloc((size_t)H * H * 2);
    u16*   w_attnW  = (u16*)  alloc((size_t)H * H * 2);
    u16*   w_decWih = (u16*)  alloc((size_t)H * (E + H) * 2);
    u16*   w_decWhh = (u16*)  alloc((size_t)H * H * 2);
    u16*   w_h2o    = (u16*)  alloc((size_t)V * 2 * H * 2);     // 131 MB, L2-resident
    u16*   emb16    = (u16*)  alloc((size_t)BN * LS * E * 2);
    float* igates   = (float*)alloc((size_t)BN * LS * H * 4);
    float* ehs32    = (float*)alloc((size_t)BN * LS * H * 4);
    u16*   ehs16    = (u16*)  alloc((size_t)BN * LS * H * 2);
    float* energy   = (float*)alloc((size_t)BN * LS * H * 4);
    u16*   h0       = (u16*)  alloc((size_t)BN * H * 2);
    float* hbuf     = (float*)alloc((size_t)BN * H * 4);
    float* tmpbuf   = (float*)alloc((size_t)BN * H * 4);
    float* scores   = (float*)alloc((size_t)BN * LS * 4);
    float* attw     = (float*)alloc((size_t)BN * LS * 4);
    u16*   xbuf     = (u16*)  alloc((size_t)BN * (E + H) * 2);
    u16*   X        = (u16*)  alloc((size_t)M2 * 2 * H * 2);    // [h_t ; ctx_t] rows

    // ---- 1) weights -> bf16 ----
    auto conv = [&](const float* in, u16* o, size_t n) {
        int blocks = (int)((n + 255) / 256);
        if (blocks > 65535) blocks = 65535;
        f32_to_bf16_kernel<<<blocks, 256, 0, stream>>>(in, o, n);
    };
    conv(enc_Wih, w_encWih, (size_t)H * E);
    conv(enc_Whh, w_encWhh, (size_t)H * H);
    conv(attn_W,  w_attnW,  (size_t)H * H);
    conv(dec_Wih, w_decWih, (size_t)H * (E + H));
    conv(dec_Whh, w_decWhh, (size_t)H * H);
    conv(h2o_W,   w_h2o,    (size_t)V * 2 * H);
    fill_zero_u16_kernel<<<256, 256, 0, stream>>>(h0, (size_t)BN * H);

    // ---- 2) encoder embedding gather + igates GEMM (4096 x 1024, K=512) ----
    enc_gather_kernel<<<4096, 256, 0, stream>>>(src, enc_embed, emb16,
                                                (size_t)BN * LS * E);
    wmma_gemm<8, 1, false, false, false, false>
        <<<dim3(H / 64, (BN * LS) / 128), 256, 0, stream>>>(
            emb16, E, w_encWih, enc_bih, nullptr, 0,
            igates, H, nullptr, 0, BN * LS, H, E);

    // ---- 3) encoder scan: h = tanh(ig_t + h @ Whh^T + bhh), 128 steps ----
    for (int t = 0; t < LS; ++t) {
        const u16* a = t ? (ehs16 + (size_t)(t - 1) * H) : h0;
        const int lda = t ? LS * H : H;
        wmma_gemm<2, 4, true, true, true, false>
            <<<dim3(H / 256, 1), 256, 0, stream>>>(
                a, lda, w_encWhh, enc_bhh,
                igates + (size_t)t * H, LS * H,
                ehs32 + (size_t)t * H, LS * H,
                ehs16 + (size_t)t * H, LS * H,
                BN, H, H);
    }

    // ---- 4) energy = enc_hs @ attn_W^T + attn_b (4096 x 1024, K=1024) ----
    wmma_gemm<8, 1, false, false, false, false>
        <<<dim3(H / 64, (BN * LS) / 128), 256, 0, stream>>>(
            ehs16, H, w_attnW, attn_b, nullptr, 0,
            energy, H, nullptr, 0, BN * LS, H, H);

    // ---- 5) decoder scan (63 steps); logits GEMM deferred ----
    for (int t = 0; t < TDEC; ++t) {
        const float* h32  = t ? hbuf : (ehs32 + (size_t)(LS - 1) * H);
        const int    ldh  = t ? H : LS * H;
        const u16*   h16  = t ? (X + (size_t)(t - 1) * 2 * H)
                              : (ehs16 + (size_t)(LS - 1) * H);
        const int    ldh16 = t ? TDEC * 2 * H : LS * H;

        scores_kernel<<<BN * LS, 256, 0, stream>>>(energy, h32, ldh, scores);
        softmax_kernel<<<BN, 128, 0, stream>>>(scores, src, attw);
        dec_gather_kernel<<<(BN * E) / 256, 256, 0, stream>>>(tgt, dec_embed,
                                                              xbuf, t);
        ctx_kernel<<<dim3(BN, H / 256), 256, 0, stream>>>(attw, ehs32, X, xbuf, t);

        // tmp = [e_t ; ctx] @ dec_Wih^T + dec_bih        (K = 1536)
        wmma_gemm<2, 4, false, false, false, false>
            <<<dim3(H / 256, 1), 256, 0, stream>>>(
                xbuf, E + H, w_decWih, dec_bih, nullptr, 0,
                tmpbuf, H, nullptr, 0, BN, H, E + H);
        // h_new = tanh(tmp + h @ dec_Whh^T + dec_bhh); bf16 mirror -> X[:, t, 0:H]
        wmma_gemm<2, 4, true, true, true, false>
            <<<dim3(H / 256, 1), 256, 0, stream>>>(
                h16, ldh16, w_decWhh, dec_bhh,
                tmpbuf, H, hbuf, H,
                X + (size_t)t * 2 * H, TDEC * 2 * H, BN, H, H);
    }

    // ---- 6) big output projection: (2016 x 2048) x (2048 x 32000) ----
    wmma_gemm<8, 1, false, false, false, true>
        <<<dim3(V / 64, (M2 + 127) / 128), 256, 0, stream>>>(
            X, 2 * H, w_h2o, h2o_b, nullptr, 0,
            out, V, nullptr, 0, M2, V, 2 * H);

    // ---- 7) in-place log_softmax over vocab ----
    log_softmax_kernel<<<M2, 256, 0, stream>>>(out);
}
